// VanillaLSTM_77558519431264
// MI455X (gfx1250) — matrix operand order repaired and checked by reading, not compiled, not run on hardware
//
#include <hip/hip_runtime.h>

// ---------------------------------------------------------------------------
// Fused 2-layer LSTM (H=128, B=8, T=102400) + mean-pool + FC1(relu) + FC2
// Persistent single-workgroup kernel for MI455X (gfx1250, wave32, WMMA).
//
// The recurrence is strictly sequential in T -> latency problem. One
// workgroup (8 waves) keeps all recurrent state in registers, layer-1
// weights (256KB f16) in LDS, layer-0 recurrent weights in VGPRs, and does
// the per-step [16 x 512 x K] GEMMs with v_wmma_f32_16x16x32_f16.
// h1/h2 are double-buffered on timestep parity so each step needs only ONE
// workgroup barrier (between layer-0 h1 writes and layer-1 h1 reads).
// Activations use the CDNA5 hardware v_tanh_f32 transcendental
// (sigmoid(x) = 0.5*tanh(0.5x)+0.5), which co-executes with WMMA/VALU.
// ---------------------------------------------------------------------------

#define H       128
#define HPAD    136              // f16 row pitch for h buffers (16B-aligned rows)
#define MDIM    16               // WMMA M tile (batch 8 padded to 16)
#define TBLK    512              // x staging block (timesteps)
#define NTHREADS 256             // 8 waves
#define NWAVES  8

typedef __attribute__((ext_vector_type(16))) _Float16 v16h;
typedef __attribute__((ext_vector_type(8)))  _Float16 v8h;
typedef __attribute__((ext_vector_type(8)))  float    v8f;

struct SMem {
  _Float16 w1[NWAVES * 32 * 32 * 16];  // 256 KB: layer-1 B fragments, pre-swizzled
  float    xs[TBLK][8];                // 16 KB: staged input scalars x[t][b]
  _Float16 h1[2][MDIM][HPAD];          // 8.5 KB: layer-1 input h, double buffered
  _Float16 h2[2][MDIM][HPAD];          // 8.5 KB: layer-2 hidden h, double buffered
  float    pooled[8][H];               // 4 KB
  float    hid[8][64];                 // 2 KB
};                                      // total ~295 KB < 320 KB LDS/WGP

// CDNA5 hardware tanh (TRANS32 op; co-executes with VALU and XDL WMMA).
__device__ __forceinline__ float tanhx(float x) {
#if __has_builtin(__builtin_amdgcn_tanhf)
  return __builtin_amdgcn_tanhf(x);
#elif __has_builtin(__builtin_amdgcn_tanh_f32)
  return __builtin_amdgcn_tanh_f32(x);
#else
  float y;
  // TRANS result hazard: 1 independent op before use (ISA 7.4) -> delay_alu.
  asm volatile("v_tanh_f32 %0, %1\n\t"
               "s_delay_alu instid0(TRANS32_DEP_1)"
               : "=v"(y) : "v"(x));
  return y;
#endif
}

// sigmoid via hw tanh: 1 trans + 2 valu instead of mul/exp/add/rcp chain.
__device__ __forceinline__ float sigf(float x) {
  return __builtin_fmaf(0.5f, tanhx(0.5f * x), 0.5f);
}

__device__ __forceinline__ v8f wmma_f16(v16h a, v16h b, v8f c) {
  // (neg_a, A, neg_b, B, c_mod, C, reuse_a, reuse_b)
  return __builtin_amdgcn_wmma_f32_16x16x32_f16(false, a, false, b, (short)0, c,
                                                false, false);
}

// Load one 16x32 A fragment (f16) from an h buffer in LDS.
// A k order used here (contiguous, not the ISA table order): lane L holds row M=L%16;
//   halves 0..7  -> K = kb + 16*(L>=16) + 0..7
//   halves 8..15 -> K = kb + 16*(L>=16) + 8 + 0..7
__device__ __forceinline__ v16h load_a_frag(const _Float16* row, int kb, int hiL) {
  const v8h lo = *(const v8h*)(row + kb + 16 * hiL);
  const v8h hi = *(const v8h*)(row + kb + 16 * hiL + 8);
  v16h a;
#pragma unroll
  for (int i = 0; i < 8; ++i) { a[i] = lo[i]; a[i + 8] = hi[i]; }
  return a;
}

// Load one 32x16 B fragment of layer-1 weights from pre-swizzled LDS:
// each lane's 16 halves are stored contiguously (32B) -> two b128 loads.
__device__ __forceinline__ v16h load_b1_frag(const _Float16* w1, int w, int g,
                                             int kf, int L) {
  const int fid = w * 32 + g * 8 + kf;
  const _Float16* p = w1 + (((fid * 32) + L) << 4);
  const v8h lo = *(const v8h*)p;
  const v8h hi = *(const v8h*)(p + 8);
  v16h b;
#pragma unroll
  for (int i = 0; i < 8; ++i) { b[i] = lo[i]; b[i + 8] = hi[i]; }
  return b;
}

__global__ __launch_bounds__(NTHREADS, 1)
void lstm_fused_kernel(const float* __restrict__ x,
                       const float* __restrict__ Wih0, const float* __restrict__ Whh0,
                       const float* __restrict__ bih0, const float* __restrict__ bhh0,
                       const float* __restrict__ Wih1, const float* __restrict__ Whh1,
                       const float* __restrict__ bih1, const float* __restrict__ bhh1,
                       const float* __restrict__ fcW1, const float* __restrict__ fcb1,
                       const float* __restrict__ fcW2, const float* __restrict__ fcb2,
                       float* __restrict__ out, int T) {
  __shared__ SMem s;

  const int tid  = threadIdx.x;
  const int w    = tid >> 5;      // wave 0..7: owns hidden units [16w, 16w+16)
  const int L    = tid & 31;      // lane
  const int lrow = L & 15;        // N column within tile / M row for A frags
  const int hiL  = L >> 4;        // lane half

  // ---- startup: layer-1 weights -> LDS as pre-swizzled f16 B fragments.
  // B layout assumed (32x16 K x N): lane L holds column N=L%16,
  //   half h -> K = 32*kf + 16*(L>=16) + h.
  // Combined K: [0,128) = Wih1, [128,256) = Whh1.
  for (int e = tid; e < NWAVES * 32 * 32 * 16; e += NTHREADS) {
    const int h    = e & 15;
    const int lane = (e >> 4) & 31;
    const int fid  = e >> 9;
    const int kf = fid & 7, g = (fid >> 3) & 3, ww = fid >> 5;
    const int n = 128 * g + 16 * ww + (lane & 15);
    const int K = 32 * kf + 16 * (lane >> 4) + h;
    const float v = (K < H) ? Wih1[n * H + K] : Whh1[n * H + (K - H)];
    s.w1[e] = (_Float16)v;
  }
  // zero both parities of the h buffers
  {
    _Float16* hz = &s.h1[0][0][0];
    for (int e = tid; e < 4 * MDIM * HPAD; e += NTHREADS) hz[e] = (_Float16)0.0f;
  }

  // ---- per-lane constants: Wih0 column (with the pad-row mask folded in,
  // so the time loop needs no cndmask) + fused biases for my 4 gate columns
  float wih0m[4], b0v[4], b1v[4];
#pragma unroll
  for (int g = 0; g < 4; ++g) {
    const int n = 128 * g + 16 * w + lrow;
    wih0m[g] = hiL ? 0.0f : Wih0[n];     // rows M>=8 are padding -> x term = 0
    b0v[g]   = bih0[n] + bhh0[n];
    b1v[g]   = bih1[n] + bhh1[n];
  }

  // ---- layer-0 recurrent weights resident in VGPRs (4 gates x 4 K-frags).
  v16h bw0[4][4];
#pragma unroll
  for (int g = 0; g < 4; ++g) {
#pragma unroll
    for (int kf = 0; kf < 4; ++kf) {
      const int n = 128 * g + 16 * w + lrow;
      const float* src = Whh0 + n * H + 32 * kf + 16 * hiL;
      v16h f;
#pragma unroll
      for (int h = 0; h < 16; ++h) f[h] = (_Float16)src[h];
      bw0[g][kf] = f;
    }
  }

  // ---- register-resident recurrent state (lane owns M = r+8*hiL, j = 16w+lrow)
  float c1[8], c2[8], pool[8];
#pragma unroll
  for (int r = 0; r < 8; ++r) { c1[r] = 0.f; c2[r] = 0.f; pool[r] = 0.f; }

  const int j = 16 * w + lrow;
  const int Mrow = 8 * hiL;       // my M rows are Mrow..Mrow+7
  __syncthreads();

  // =======================  sequential time loop  =======================
  // Per step: ONE barrier (h1[p] writes -> h1[p] reads). All other W/R pairs
  // are separated by that same barrier of the adjacent step thanks to the
  // parity double buffering of h1/h2 (see analysis in header comment).
  for (int t = 0; t < T; ++t) {
    const int p  = t & 1;
    const int tloc = t & (TBLK - 1);
    if (tloc == 0) {
      // refill x staging block [t, t+TBLK): coalesced along t per batch row,
      // and prefetch the following block into cache (global_prefetch_b8).
      const int b = tid >> 5, tl = tid & 31;
#pragma unroll
      for (int m = 0; m < TBLK / 32; ++m) {
        const int tt = t + tl + 32 * m;
        s.xs[tl + 32 * m][b] = (tt < T) ? x[(long long)b * T + tt] : 0.f;
        if (tt + TBLK < T)
          __builtin_prefetch(&x[(long long)b * T + tt + TBLK], 0, 1);
      }
      __syncthreads();
    }

    // ---- layer 0: gates0 = x*Wih0 + bias + h1[t-1] @ Whh0^T
    float xv[8];
    {
      const float4* xp = (const float4*)&s.xs[tloc][0];  // broadcast reads
      const float4 xa = xp[0], xb = xp[1];
      xv[0] = xa.x; xv[1] = xa.y; xv[2] = xa.z; xv[3] = xa.w;
      xv[4] = xb.x; xv[5] = xb.y; xv[6] = xb.z; xv[7] = xb.w;
    }
    v16h a0[4];
#pragma unroll
    for (int kf = 0; kf < 4; ++kf)
      a0[kf] = load_a_frag(&s.h1[p ^ 1][lrow][0], 32 * kf, hiL);

    v8f acc[4];
#pragma unroll
    for (int g = 0; g < 4; ++g) {
#pragma unroll
      for (int r = 0; r < 8; ++r)
        acc[g][r] = __builtin_fmaf(xv[r], wih0m[g], b0v[g]);
    }
#pragma unroll
    for (int g = 0; g < 4; ++g) {
#pragma unroll
      for (int kf = 0; kf < 4; ++kf)
        acc[g] = wmma_f16(a0[kf], bw0[g][kf], acc[g]);
    }
    // activations + cell update (all in registers; lane owns i/f/g/o of (M,j))
    float h1n[8];
#pragma unroll
    for (int r = 0; r < 8; ++r) {
      const float iv = sigf(acc[0][r]);
      const float fv = sigf(acc[1][r]);
      const float gv = tanhx(acc[2][r]);
      const float ov = sigf(acc[3][r]);
      const float c  = __builtin_fmaf(fv, c1[r], iv * gv);
      c1[r] = c;
      h1n[r] = ov * tanhx(c);
    }
#pragma unroll
    for (int r = 0; r < 8; ++r)
      s.h1[p][Mrow + r][j] = (_Float16)h1n[r];

    __syncthreads();  // the one per-step barrier: h1[p] now holds h1[t]

    // ---- layer 1: gates1 = bias + [h1[t], h2[t-1]] @ [Wih1; Whh1]^T  (K=256)
    v16h a1[4], ah2[4];
#pragma unroll
    for (int kf = 0; kf < 4; ++kf) {
      a1[kf]  = load_a_frag(&s.h1[p][lrow][0], 32 * kf, hiL);
      ah2[kf] = load_a_frag(&s.h2[p ^ 1][lrow][0], 32 * kf, hiL);
    }

    v8f acc1[4];
#pragma unroll
    for (int g = 0; g < 4; ++g) {
      v8f a;
#pragma unroll
      for (int r = 0; r < 8; ++r) a[r] = b1v[g];
#pragma unroll
      for (int kf = 0; kf < 4; ++kf)
        a = wmma_f16(a1[kf], load_b1_frag(s.w1, w, g, kf, L), a);
#pragma unroll
      for (int kf = 4; kf < 8; ++kf)
        a = wmma_f16(ah2[kf - 4], load_b1_frag(s.w1, w, g, kf, L), a);
      acc1[g] = a;
    }
    float h2n[8];
#pragma unroll
    for (int r = 0; r < 8; ++r) {
      const float iv = sigf(acc1[0][r]);
      const float fv = sigf(acc1[1][r]);
      const float gv = tanhx(acc1[2][r]);
      const float ov = sigf(acc1[3][r]);
      const float c  = __builtin_fmaf(fv, c2[r], iv * gv);
      c2[r] = c;
      const float hh = ov * tanhx(c);
      h2n[r] = hh;
      pool[r] += hh;              // on-the-fly mean-pool accumulation
    }
#pragma unroll
    for (int r = 0; r < 8; ++r)
      s.h2[p][Mrow + r][j] = (_Float16)h2n[r];
    // no end-of-step barrier needed: next step's reads hit the other parity,
    // and every conflicting access is separated by an adjacent step's barrier.
  }

  __syncthreads();  // drain last h2 stores before reusing nothing; cheap + safe

  // =======================  epilogue: pool + FCs  =======================
  const float invT = 1.0f / (float)T;
  if (!hiL) {  // rows M=0..7 are the real batches
#pragma unroll
    for (int r = 0; r < 8; ++r) s.pooled[r][j] = pool[r] * invT;
  }
  __syncthreads();

  for (int u0 = tid; u0 < 8 * 64; u0 += NTHREADS) {
    const int b = u0 >> 6, u = u0 & 63;
    float a = fcb1[u];
#pragma unroll 8
    for (int k = 0; k < H; ++k) a += fcW1[u * H + k] * s.pooled[b][k];
    s.hid[b][u] = fmaxf(a, 0.f);
  }
  __syncthreads();

  if (tid < 88) {
    const int b = tid / 11, o = tid % 11;
    float a = fcb2[o];
#pragma unroll 8
    for (int u = 0; u < 64; ++u) a += fcW2[o * 64 + u] * s.hid[b][u];
    out[b * 11 + o] = a;
  }
}

extern "C" void kernel_launch(void* const* d_in, const int* in_sizes, int n_in,
                              void* d_out, int out_size, void* d_ws, size_t ws_size,
                              hipStream_t stream) {
  const float* x    = (const float*)d_in[0];
  const float* Wih0 = (const float*)d_in[1];
  const float* Whh0 = (const float*)d_in[2];
  const float* bih0 = (const float*)d_in[3];
  const float* bhh0 = (const float*)d_in[4];
  const float* Wih1 = (const float*)d_in[5];
  const float* Whh1 = (const float*)d_in[6];
  const float* bih1 = (const float*)d_in[7];
  const float* bhh1 = (const float*)d_in[8];
  const float* fcW1 = (const float*)d_in[9];
  const float* fcb1 = (const float*)d_in[10];
  const float* fcW2 = (const float*)d_in[11];
  const float* fcb2 = (const float*)d_in[12];
  float* out = (float*)d_out;

  const int T = in_sizes[0] / 8;  // x: [8, 1, T]

  lstm_fused_kernel<<<1, NTHREADS, 0, stream>>>(
      x, Wih0, Whh0, bih0, bhh0, Wih1, Whh1, bih1, bhh1,
      fcW1, fcb1, fcW2, fcb2, out, T);
}
